// SparseMoE_35957466202707
// MI455X (gfx1250) — matrix operand order repaired and checked by reading, not compiled, not run on hardware
//
#include <hip/hip_runtime.h>
#include <hip/hip_bf16.h>

// Problem constants (from reference): B=8, S=4096, H=256, F=512, E=16, K=2
#define NTOK 32768
#define HDIM 256
#define FDIM 512
#define NEXP 16
#define CAP  8192          // per-expert token-list capacity (expected ~4096)
#define MAXB (CAP / 128)   // 64 blocks of 128 rows per expert
#define EPSV 1e-5f

typedef __attribute__((ext_vector_type(16))) __bf16 v16bf;
typedef __attribute__((ext_vector_type(8)))  float  v8f;
typedef __attribute__((ext_vector_type(4)))  unsigned int ui4;
typedef __attribute__((ext_vector_type(8)))  int i8v;
typedef __attribute__((ext_vector_type(4)))  int i4v;

union Frag {
  v16bf v;
  uint4 q[2];
};

__device__ __forceinline__ unsigned short f2bf(float f) {
  unsigned int u = __float_as_uint(f);
  u += 0x7fffu + ((u >> 16) & 1u);   // round-to-nearest-even
  return (unsigned short)(u >> 16);
}

// A-matrix 16x32 bf16 fragment (ISA 7.12.2): lane&15 = M row;
// lanes 0-15 hold K {kb..kb+7, kb+16..kb+23}, lanes 16-31 hold {kb+8..15, kb+24..31}.
__device__ __forceinline__ v16bf load_a_frag(const unsigned short* base, int stride,
                                             int m0, int kb, int lane) {
  const unsigned short* p = base + (m0 + (lane & 15)) * stride + kb + ((lane >> 4) << 3);
  Frag f;
  f.q[0] = *(const uint4*)(p);
  f.q[1] = *(const uint4*)(p + 16);
  return f.v;
}

// B-matrix 32x16 bf16 fragment from a transposed [N][K] LDS panel:
// lane&15 = N col; lanes 0-15 hold K {kb..kb+7, kb+16..kb+23}, lanes 16-31 hold {kb+8..15, kb+24..31}.
__device__ __forceinline__ v16bf load_b_frag(const unsigned short* baseT, int stride,
                                             int n0, int kb, int lane) {
  const unsigned short* p = baseT + (n0 + (lane & 15)) * stride + kb + ((lane >> 4) << 3);
  Frag f;
  f.q[0] = *(const uint4*)(p);
  f.q[1] = *(const uint4*)(p + 16);
  return f.v;
}

// Issue a TDM 2D-tile load (bf16 elements) of [rows x width] from global into
// LDS at lds_byte_off, appending pad_amount=4 DWORDs after every pad_interval
// DWORDs (ISA 8.3/8.4 D# layout; data_size code 1 -> 2 bytes; type=2).
// 6-arg builtin form (clang-23): (g0 u32x4, g1 i32x8, g2 i32x4, g3 i32x4,
// g4 i32x8, cpol).
__device__ __forceinline__ void tdm_load_2d_bf16(unsigned lds_byte_off,
                                                 const unsigned short* gsrc,
                                                 unsigned width, unsigned rows,
                                                 unsigned pad_interval_code) {
  unsigned long long ga = (unsigned long long)(uintptr_t)gsrc;
  ui4 g0;
  g0[0] = 1u;                                        // count=1, no gather
  g0[1] = lds_byte_off;                              // lds_addr
  g0[2] = (unsigned)(ga & 0xFFFFFFFFull);            // global_addr[31:0]
  g0[3] = (unsigned)((ga >> 32) & 0x01FFFFFFull)     // global_addr[56:32]
          | (2u << 30);                              // type = 2
  i8v g1;
  g1[0] = (int)((1u << 16)                           // data_size = 2 B
          | (1u << 20)                               // pad_enable
          | (pad_interval_code << 22)                // pad interval code
          | (3u << 25));                             // pad_amount = 4 DWORDs
  g1[1] = (int)((width & 0xFFFFu) << 16);            // tensor_dim0[15:0]
  g1[2] = (int)(((width >> 16) & 0xFFFFu)            // tensor_dim0[31:16]
          | ((rows & 0xFFFFu) << 16));               // tensor_dim1[15:0]
  g1[3] = (int)(((rows >> 16) & 0xFFFFu)             // tensor_dim1[31:16]
          | ((width & 0xFFFFu) << 16));              // tile_dim0
  g1[4] = (int)(rows & 0xFFFFu);                     // tile_dim1 (tile_dim2 = 0)
  g1[5] = (int)width;                                // tensor_dim0_stride[31:0]
  g1[6] = 0;                                         // stride0 hi / stride1 lo
  g1[7] = 0;
  i4v g2 = {0, 0, 0, 0};                             // dims 2/3 unused (2D tile)
  i4v g3 = {0, 0, 0, 0};
  i8v g4 = {0, 0, 0, 0, 0, 0, 0, 0};
  __builtin_amdgcn_tensor_load_to_lds(g0, g1, g2, g3, g4, 0);
}

// ---------------------------------------------------------------------------
// Pass 0: weight transpose + fp32->bf16 convert (once per launch, L2-resident).
// src: [E][R][C] fp32  ->  dst: [E][C][R] bf16. LDS-tiled, coalesced both sides.
// ---------------------------------------------------------------------------
__global__ void transpose_bf16_kernel(const float* __restrict__ src,
                                      unsigned short* __restrict__ dst,
                                      int R, int C) {
  __shared__ float tile[32][33];
  const int e  = blockIdx.z;
  const int r0 = blockIdx.y * 32;
  const int c0 = blockIdx.x * 32;
  const int tx  = threadIdx.x & 31;
  const int ty4 = (threadIdx.x >> 5) << 2;
  const float* s = src + (size_t)e * R * C;
  unsigned short* d = dst + (size_t)e * R * C;
#pragma unroll
  for (int i = 0; i < 4; ++i)
    tile[ty4 + i][tx] = s[(size_t)(r0 + ty4 + i) * C + (c0 + tx)];
  __syncthreads();
#pragma unroll
  for (int i = 0; i < 4; ++i)
    d[(size_t)(c0 + ty4 + i) * R + (r0 + tx)] = f2bf(tile[tx][ty4 + i]);
}

// ---------------------------------------------------------------------------
// Pass 1: gating. One wave32 per token: 16 expert logits, top-2, softmax,
// atomic append (token, weight) into per-expert lists.
// ---------------------------------------------------------------------------
__global__ void moe_gate_kernel(const float* __restrict__ x,
                                const float* __restrict__ gw,
                                const float* __restrict__ gb,
                                int* __restrict__ counts,
                                int* __restrict__ toks,
                                float* __restrict__ wts) {
  const int wid  = (blockIdx.x * blockDim.x + threadIdx.x) >> 5;
  const int lane = threadIdx.x & 31;
  if (wid >= NTOK) return;

  float acc[NEXP];
#pragma unroll
  for (int e = 0; e < NEXP; ++e) acc[e] = 0.f;

  const float* xrow = x + (size_t)wid * HDIM;
#pragma unroll
  for (int i = 0; i < HDIM / 32; ++i) {
    const int h = lane + i * 32;
    const float xv = xrow[h];
    const float4* g4 = (const float4*)(gw + (size_t)h * NEXP);
#pragma unroll
    for (int q = 0; q < 4; ++q) {
      const float4 g = g4[q];
      acc[q * 4 + 0] += xv * g.x;
      acc[q * 4 + 1] += xv * g.y;
      acc[q * 4 + 2] += xv * g.z;
      acc[q * 4 + 3] += xv * g.w;
    }
  }
#pragma unroll
  for (int off = 16; off > 0; off >>= 1)
#pragma unroll
    for (int e = 0; e < NEXP; ++e) acc[e] += __shfl_xor(acc[e], off, 32);

  if (lane == 0) {
#pragma unroll
    for (int e = 0; e < NEXP; ++e) acc[e] += gb[e];
    int i1 = 0; float v1 = acc[0];
#pragma unroll
    for (int e = 1; e < NEXP; ++e)
      if (acc[e] > v1) { v1 = acc[e]; i1 = e; }
    int i2 = -1; float v2 = -3.4e38f;
#pragma unroll
    for (int e = 0; e < NEXP; ++e)
      if (e != i1 && acc[e] > v2) { v2 = acc[e]; i2 = e; }
    const float e2  = __expf(v2 - v1);
    const float inv = 1.f / (1.f + e2);
    const float w1v = inv;          // softmax over {v1, v2}
    const float w2v = e2 * inv;

    int p1 = atomicAdd(&counts[i1], 1);
    if (p1 < CAP) { toks[i1 * CAP + p1] = wid; wts[i1 * CAP + p1] = w1v; }
    int p2 = atomicAdd(&counts[i2], 1);
    if (p2 < CAP) { toks[i2 * CAP + p2] = wid; wts[i2 * CAP + p2] = w2v; }
  }
}

// ---------------------------------------------------------------------------
// Pass 2: routed expert FFN via bf16 WMMA; weight panels staged by the TDM.
// Block = (expert e, 128-row token block). 8 waves; wave w owns M-tile rows
// [w*16, w*16+16).
// LDS layout (dynamic, 268288 B):
//   sX  [128][264] bf16    (row 528 B, 16B aligned)
//   sH  [128][520] bf16    (row 1040 B)
//   sW  [64][264] / [64][520] bf16 (TDM-staged transposed weight panel)
//   sTok[128], sWt[128]
// ---------------------------------------------------------------------------
__global__ __launch_bounds__(256, 1) void moe_ffn_kernel(
    const float* __restrict__ x,
    const unsigned short* __restrict__ w1t,   // [E][F][H] bf16
    const float* __restrict__ b1,
    const unsigned short* __restrict__ w2t,   // [E][H][F] bf16
    const float* __restrict__ b2,
    const int* __restrict__ counts,
    const int* __restrict__ toks, const float* __restrict__ wts,
    float* __restrict__ out) {
  extern __shared__ unsigned char smem[];
  unsigned short* sX  = (unsigned short*)smem;                            // 67584 B
  unsigned short* sH  = (unsigned short*)(smem + 67584);                  // 133120 B
  unsigned short* sW  = (unsigned short*)(smem + 67584 + 133120);         // 66560 B
  int*            sTok = (int*)(smem + 67584 + 133120 + 66560);           // 512 B
  float*          sWt  = (float*)(smem + 67584 + 133120 + 66560 + 512);   // 512 B
  const unsigned sW_lds = __builtin_amdgcn_groupstaticsize() + 200704u;   // raw LDS byte addr of sW

  const int e = blockIdx.x;
  int cnt = counts[e];
  if (cnt > CAP) cnt = CAP;
  const int base = blockIdx.y * 128;
  if (base >= cnt) return;                     // uniform early exit
  const int nrows = min(128, cnt - base);

  const int tid  = threadIdx.x;
  const int lane = tid & 31;
  const int wv   = tid >> 5;
  const int m0   = wv * 16;

  if (tid < 128) {
    const int i = base + tid;
    if (i < cnt) { sTok[tid] = toks[e * CAP + i]; sWt[tid] = wts[e * CAP + i]; }
    else         { sTok[tid] = 0;                 sWt[tid] = 0.f; }
  }
  __syncthreads();

  // Gather X rows (fp32 -> bf16) into LDS.
  for (int i = tid; i < 128 * 64; i += 256) {
    const int row = i >> 6;
    const int c4  = (i & 63) << 2;
    const float4 v = *(const float4*)(x + (size_t)sTok[row] * HDIM + c4);
    ushort4 h;
    h.x = f2bf(v.x); h.y = f2bf(v.y); h.z = f2bf(v.z); h.w = f2bf(v.w);
    *(ushort4*)(sX + row * 264 + c4) = h;
  }
  __syncthreads();

  // ---- GEMM1: sH = relu(sX @ W1 + b1), N = 512 in 8 panels of 64 ----
  for (int fp = 0; fp < 8; ++fp) {
    // TDM: stage W1^T panel [64 f][256 k] bf16; 128-DW rows + 4-DW pad -> stride 264.
    if (wv == 0) {
      tdm_load_2d_bf16(sW_lds, w1t + ((size_t)e * FDIM + fp * 64) * HDIM,
                       HDIM, 64u, 6u /* 128 DWORDs */);
      __builtin_amdgcn_s_wait_tensorcnt(0);
    }
    __syncthreads();

    v8f acc[4] = {};
#pragma unroll
    for (int kc = 0; kc < 8; ++kc) {
      const v16bf a = load_a_frag(sX, 264, m0, kc * 32, lane);
#pragma unroll
      for (int j = 0; j < 4; ++j) {
        const v16bf b = load_b_frag(sW, 264, j * 16, kc * 32, lane);
        acc[j] = __builtin_amdgcn_wmma_f32_16x16x32_bf16(
            false, a, false, b, (short)0, acc[j], false, false);
      }
    }
    // Epilogue: +b1, relu, bf16 -> sH. C layout: lane 0-15 -> M=r, lane 16-31 -> M=r+8.
#pragma unroll
    for (int j = 0; j < 4; ++j) {
      const int f = fp * 64 + j * 16 + (lane & 15);
      const float bias = b1[e * FDIM + f];
#pragma unroll
      for (int r = 0; r < 8; ++r) {
        const int m = m0 + ((lane >> 4) << 3) + r;
        float v = acc[j][r] + bias;
        v = v > 0.f ? v : 0.f;
        sH[m * 520 + f] = f2bf(v);
      }
    }
    __syncthreads();
  }

  // ---- GEMM2: O = sH @ W2 + b2, N = 256 in 4 panels of 64 ----
  for (int hp = 0; hp < 4; ++hp) {
    // TDM: stage W2^T panel [64 h][512 k] bf16; 256-DW rows + 4-DW pad -> stride 520.
    if (wv == 0) {
      tdm_load_2d_bf16(sW_lds, w2t + ((size_t)e * HDIM + hp * 64) * FDIM,
                       FDIM, 64u, 7u /* 256 DWORDs */);
      __builtin_amdgcn_s_wait_tensorcnt(0);
    }
    __syncthreads();

    v8f acc[4] = {};
#pragma unroll
    for (int kc = 0; kc < 16; ++kc) {
      const v16bf a = load_a_frag(sH, 520, m0, kc * 32, lane);
#pragma unroll
      for (int j = 0; j < 4; ++j) {
        const v16bf b = load_b_frag(sW, 520, j * 16, kc * 32, lane);
        acc[j] = __builtin_amdgcn_wmma_f32_16x16x32_bf16(
            false, a, false, b, (short)0, acc[j], false, false);
      }
    }
    // Epilogue: +b2, * gate weight, atomic scatter into out.
#pragma unroll
    for (int j = 0; j < 4; ++j) {
      const int h = hp * 64 + j * 16 + (lane & 15);
      const float bias = b2[e * HDIM + h];
#pragma unroll
      for (int r = 0; r < 8; ++r) {
        const int m = m0 + ((lane >> 4) << 3) + r;
        if (m < nrows) {
          atomicAdd(out + (size_t)sTok[m] * HDIM + h, (acc[j][r] + bias) * sWt[m]);
        }
      }
    }
    __syncthreads();
  }
}

// ---------------------------------------------------------------------------
// Pass 3: residual + LayerNorm (in-place on out). One wave32 per token.
// ---------------------------------------------------------------------------
__global__ void moe_ln_kernel(const float* __restrict__ x,
                              const float* __restrict__ gamma,
                              const float* __restrict__ beta,
                              float* __restrict__ out) {
  const int wid  = (blockIdx.x * blockDim.x + threadIdx.x) >> 5;
  const int lane = threadIdx.x & 31;
  if (wid >= NTOK) return;

  const size_t base = (size_t)wid * HDIM + lane * 8;
  float y[8];
  float s = 0.f, ss = 0.f;
#pragma unroll
  for (int i = 0; i < 8; ++i) {
    const float v = x[base + i] + out[base + i];
    y[i] = v;
    s += v;
    ss += v * v;
  }
#pragma unroll
  for (int off = 16; off > 0; off >>= 1) {
    s  += __shfl_xor(s, off, 32);
    ss += __shfl_xor(ss, off, 32);
  }
  const float mu   = s * (1.f / HDIM);
  const float var  = ss * (1.f / HDIM) - mu * mu;
  const float rstd = rsqrtf(var + EPSV);
#pragma unroll
  for (int i = 0; i < 8; ++i) {
    const int h = lane * 8 + i;
    out[base + i] = (y[i] - mu) * rstd * gamma[h] + beta[h];
  }
}

// ---------------------------------------------------------------------------
extern "C" void kernel_launch(void* const* d_in, const int* in_sizes, int n_in,
                              void* d_out, int out_size, void* d_ws, size_t ws_size,
                              hipStream_t stream) {
  const float* x      = (const float*)d_in[0];
  const float* gate_w = (const float*)d_in[1];
  const float* gate_b = (const float*)d_in[2];
  const float* w1     = (const float*)d_in[3];
  const float* b1     = (const float*)d_in[4];
  const float* w2     = (const float*)d_in[5];
  const float* b2     = (const float*)d_in[6];
  const float* gamma  = (const float*)d_in[7];
  const float* beta   = (const float*)d_in[8];
  float* out = (float*)d_out;

  // Workspace layout:
  //   [0, 4096)            counts (16 ints, padded)
  //   [4096, +512K)        toks
  //   [+512K, +1M)         wts
  //   [w1t, 4 MB)          w1^T bf16 [E][F][H]
  //   [w2t, 4 MB)          w2^T bf16 [E][H][F]
  char* ws = (char*)d_ws;
  int*            counts = (int*)ws;
  int*            toks   = (int*)(ws + 4096);
  float*          wts    = (float*)(ws + 4096 + sizeof(int) * NEXP * CAP);
  unsigned short* w1t    = (unsigned short*)(ws + 4096 + 2 * sizeof(int) * NEXP * CAP);
  unsigned short* w2t    = w1t + (size_t)NEXP * FDIM * HDIM;

  (void)hipMemsetAsync(counts, 0, 256, stream);
  (void)hipMemsetAsync(out, 0, sizeof(float) * (size_t)NTOK * HDIM, stream);

  // Weight transpose + bf16 convert: w1 [E][H][F] -> w1t [E][F][H]; w2 likewise.
  transpose_bf16_kernel<<<dim3(FDIM / 32, HDIM / 32, NEXP), 256, 0, stream>>>(w1, w1t, HDIM, FDIM);
  transpose_bf16_kernel<<<dim3(HDIM / 32, FDIM / 32, NEXP), 256, 0, stream>>>(w2, w2t, FDIM, HDIM);

  moe_gate_kernel<<<NTOK / 8, 256, 0, stream>>>(x, gate_w, gate_b, counts, toks, wts);

  const size_t smem_bytes = 67584 + 133120 + 66560 + 512 + 512; // 268288 B
  moe_ffn_kernel<<<dim3(NEXP, MAXB), 256, smem_bytes, stream>>>(
      x, w1t, b1, w2t, b2, counts, toks, wts, out);

  moe_ln_kernel<<<NTOK / 8, 256, 0, stream>>>(x, gamma, beta, out);
}
